// SparseInterpolatedExperts_60928406061673
// MI455X (gfx1250) — matrix operand order repaired and checked by reading, not compiled, not run on hardware
//
#include <hip/hip_runtime.h>
#include <hip/hip_bf16.h>

// ---------------- constants ----------------
#define BB 4
#define NN 256
#define DD 384
#define TT 260            // 4 skill tokens + 256
#define HH 4
#define DH 96
#define LL 12
#define FF1 1536
#define FF2 384
#define MM 8
#define NLOC ((size_t)LL * FF1 * FF2)          // 7,077,888 per tensor
#define OUT_BSTRIDE (2ull * NLOC)              // per-batch output stride

typedef __attribute__((ext_vector_type(16))) __bf16 v16bf;
typedef __attribute__((ext_vector_type(8)))  __bf16 v8bf;
typedef __attribute__((ext_vector_type(8)))  float  v8f;
typedef __attribute__((ext_vector_type(4)))  float  f4;

// ---------------- WMMA bf16 GEMM: C[M,N] = A[M,K] @ B[K,N] + bias ----------------
// Block = 8 waves = a 128-row band x 64-col strip. Each wave owns one 16-row
// m-tile and computes a 16x64 strip: one A fragment reused across 4 WMMAs per
// K-step. The 32x64 B panel per K-step is staged once per block into LDS,
// COLUMN-MAJOR (bs[col][k]) with a 40-element (80 B = 5x16 B) row pitch so each
// lane's fragment is two aligned, bank-conflict-free ds_load_b128's.
// K % 32 == 0, N % 64 == 0. M handled by clamping the last band (wave-uniform
// store guard keeps EXEC all-ones for WMMA).
__global__ void gemm_bf16_wmma(const float* __restrict__ A, const float* __restrict__ Bm,
                               const float* __restrict__ bias, float* __restrict__ C,
                               int Mrows, int Ncols, int K) {
  const int lane  = threadIdx.x & 31;
  const int wave  = threadIdx.x >> 5;
  const int nStrips = Ncols >> 6;
  const int band  = blockIdx.x / nStrips;
  const int strip = blockIdx.x - band * nStrips;
  const int mTiles = Mrows >> 4;
  int mt = band * 8 + wave;
  const bool valid = (mt < mTiles);            // wave-uniform
  if (!valid) mt = mTiles - 1;                 // clamp: redundant compute, no store
  const int lan15 = lane & 15;
  const int hi    = lane >> 4;
  const int rowA  = (mt << 4) + lan15;
  const int colBase = strip << 6;

  __shared__ __bf16 bs[64][40];                // [col][k], pitch 80B (16B aligned)

  v8f acc0 = {}, acc1 = {}, acc2 = {}, acc3 = {};
  for (int k0 = 0; k0 < K; k0 += 32) {
    __syncthreads();
    // cooperative fill: 2048 elems / 256 threads = 8 each; global reads
    // coalesced over columns, transposed bf16 stores into LDS
#pragma unroll
    for (int i = 0; i < 8; ++i) {
      int idx = threadIdx.x + i * 256;
      int kr = idx >> 6, c = idx & 63;
      bs[c][kr] = (__bf16)Bm[(size_t)(k0 + kr) * Ncols + colBase + c];
    }
    __syncthreads();
    // A fragment: lane holds row `rowA`, K = k0 + hi*8 + {0..7} and {16..23}
    const float* ap = A + (size_t)rowA * K + k0 + hi * 8;
    f4 a0 = *(const f4*)(ap);
    f4 a1 = *(const f4*)(ap + 4);
    f4 a2 = *(const f4*)(ap + 16);
    f4 a3 = *(const f4*)(ap + 20);
    v16bf af;
    af[0]=(__bf16)a0.x; af[1]=(__bf16)a0.y; af[2]=(__bf16)a0.z; af[3]=(__bf16)a0.w;
    af[4]=(__bf16)a1.x; af[5]=(__bf16)a1.y; af[6]=(__bf16)a1.z; af[7]=(__bf16)a1.w;
    af[8]=(__bf16)a2.x; af[9]=(__bf16)a2.y; af[10]=(__bf16)a2.z; af[11]=(__bf16)a2.w;
    af[12]=(__bf16)a3.x; af[13]=(__bf16)a3.y; af[14]=(__bf16)a3.z; af[15]=(__bf16)a3.w;
    // 4 B fragments from LDS: per fragment two aligned b128 loads (K = kb+{0..7}
    // and kb+16+{0..7} for this lane's column), 4 WMMAs reusing the A fragment
    const int kb = hi << 3;                    // lanes>=16 take K rows 8..15 and 24..31
#pragma unroll
    for (int t = 0; t < 4; ++t) {
      const int col = (t << 4) + lan15;
      v8bf lo = *(const v8bf*)(&bs[col][kb]);
      v8bf hi8 = *(const v8bf*)(&bs[col][kb + 16]);
      v16bf bf = __builtin_shufflevector(lo, hi8, 0, 1, 2, 3, 4, 5, 6, 7,
                                         8, 9, 10, 11, 12, 13, 14, 15);
      switch (t) {
        case 0: acc0 = __builtin_amdgcn_wmma_f32_16x16x32_bf16(false, af, false, bf, (short)0, acc0, false, false); break;
        case 1: acc1 = __builtin_amdgcn_wmma_f32_16x16x32_bf16(false, af, false, bf, (short)0, acc1, false, false); break;
        case 2: acc2 = __builtin_amdgcn_wmma_f32_16x16x32_bf16(false, af, false, bf, (short)0, acc2, false, false); break;
        case 3: acc3 = __builtin_amdgcn_wmma_f32_16x16x32_bf16(false, af, false, bf, (short)0, acc3, false, false); break;
      }
    }
  }
  if (valid) {
    const int rbase = (mt << 4) + hi * 8;      // C rows for this lane half
#pragma unroll
    for (int t = 0; t < 4; ++t) {
      const int col = colBase + (t << 4) + lan15;
      const float bv = bias[col];
      const v8f* accp = (t == 0) ? &acc0 : (t == 1) ? &acc1 : (t == 2) ? &acc2 : &acc3;
#pragma unroll
      for (int r = 0; r < 8; ++r)
        C[(size_t)(rbase + r) * Ncols + col] = (*accp)[r] + bv;
    }
  }
}

// ---------------- tok assembly: skills | h + count_enc ----------------
__global__ void assemble_tok(const float* __restrict__ h, const float* __restrict__ skills,
                             const float* __restrict__ cenc, float* __restrict__ tok) {
  int idx = blockIdx.x * 256 + threadIdx.x;
  if (idx >= BB * TT * DD) return;
  int col = idx % DD;
  int rowt = idx / DD;
  int b = rowt / TT, t = rowt % TT;
  float v;
  if (t < 4) v = skills[t * DD + col];
  else       v = h[((size_t)b * NN + (t - 4)) * DD + col] + cenc[col];
  tok[idx] = v;
}

// ---------------- LayerNorm (one row per block, 128 threads) ----------------
__global__ void ln_kernel(const float* __restrict__ in, float* __restrict__ out,
                          const float* __restrict__ g, const float* __restrict__ bb,
                          int remap) {
  int row = blockIdx.x;
  int src = remap ? ((row >> 2) * TT + (row & 3)) : row;   // remap: 16 hot rows
  const float* xr = in + (size_t)src * DD;
  __shared__ float red[128];
  int tid = threadIdx.x;
  float s = 0.f;
  for (int i = tid; i < DD; i += 128) s += xr[i];
  red[tid] = s; __syncthreads();
  for (int off = 64; off > 0; off >>= 1) { if (tid < off) red[tid] += red[tid + off]; __syncthreads(); }
  float mean = red[0] * (1.f / DD);
  __syncthreads();
  float v = 0.f;
  for (int i = tid; i < DD; i += 128) { float d = xr[i] - mean; v += d * d; }
  red[tid] = v; __syncthreads();
  for (int off = 64; off > 0; off >>= 1) { if (tid < off) red[tid] += red[tid + off]; __syncthreads(); }
  float rstd = rsqrtf(red[0] * (1.f / DD) + 1e-6f);
  for (int i = tid; i < DD; i += 128)
    out[(size_t)row * DD + i] = (xr[i] - mean) * rstd * g[i] + bb[i];
}

// ---------------- attention: only 4 query tokens per (b,h) ----------------
__global__ void attn_kernel(const float* __restrict__ qkv, float* __restrict__ o4) {
  const int b = blockIdx.x >> 2, h = blockIdx.x & 3;
  const int ROW = 3 * DD;
  __shared__ float qs[4 * DH];
  __shared__ float p[4 * TT];
  __shared__ float red[256];
  const float* base = qkv + (size_t)b * TT * ROW;
  for (int idx = threadIdx.x; idx < 4 * DH; idx += 256) {
    int t = idx / DH, d = idx % DH;
    qs[idx] = base[(size_t)t * ROW + h * DH + d];              // q slice
  }
  __syncthreads();
  for (int k = threadIdx.x; k < TT; k += 256) {
    const float* kv = base + (size_t)k * ROW + DD + h * DH;    // k slice
    float a0 = 0, a1 = 0, a2 = 0, a3 = 0;
    for (int d = 0; d < DH; ++d) {
      float kd = kv[d];
      a0 += qs[d] * kd; a1 += qs[DH + d] * kd;
      a2 += qs[2 * DH + d] * kd; a3 += qs[3 * DH + d] * kd;
    }
    const float sc = 0.1020620726f;                            // 1/sqrt(96)
    p[k] = a0 * sc; p[TT + k] = a1 * sc; p[2 * TT + k] = a2 * sc; p[3 * TT + k] = a3 * sc;
  }
  __syncthreads();
  for (int t = 0; t < 4; ++t) {
    float m = -1e30f;
    for (int k = threadIdx.x; k < TT; k += 256) m = fmaxf(m, p[t * TT + k]);
    red[threadIdx.x] = m; __syncthreads();
    for (int off = 128; off > 0; off >>= 1) { if (threadIdx.x < off) red[threadIdx.x] = fmaxf(red[threadIdx.x], red[threadIdx.x + off]); __syncthreads(); }
    m = red[0]; __syncthreads();
    float s = 0.f;
    for (int k = threadIdx.x; k < TT; k += 256) { float e = __expf(p[t * TT + k] - m); p[t * TT + k] = e; s += e; }
    red[threadIdx.x] = s; __syncthreads();
    for (int off = 128; off > 0; off >>= 1) { if (threadIdx.x < off) red[threadIdx.x] += red[threadIdx.x + off]; __syncthreads(); }
    float inv = 1.f / red[0]; __syncthreads();
    for (int k = threadIdx.x; k < TT; k += 256) p[t * TT + k] *= inv;
    __syncthreads();
  }
  for (int idx = threadIdx.x; idx < 4 * DH; idx += 256) {
    int t = idx / DH, d = idx % DH;
    const float* vb = base + 2 * DD + h * DH + d;              // v slice
    float acc = 0.f;
    for (int k = 0; k < TT; ++k) acc += p[t * TT + k] * vb[(size_t)k * ROW];
    o4[((size_t)b * 4 + t) * DD + h * DH + d] = acc;
  }
}

// ---------------- o-projection on 16 hot rows: tok += o4 @ Wo + bo ----------------
__global__ void oproj_kernel(const float* __restrict__ o4, const float* __restrict__ Wo,
                             const float* __restrict__ bo, float* __restrict__ tok) {
  int idx = blockIdx.x * 256 + threadIdx.x;
  if (idx >= 16 * DD) return;
  int row = idx / DD, col = idx % DD;
  const float* a = o4 + (size_t)row * DD;
  float s = bo[col];
  for (int d = 0; d < DD; ++d) s += a[d] * Wo[(size_t)d * DD + col];
  int b = row >> 2, t = row & 3;
  tok[((size_t)(b * TT + t)) * DD + col] += s;
}

// ---------------- FFN on 16 hot rows ----------------
__global__ void ffn1_kernel(const float* __restrict__ y2, const float* __restrict__ W1,
                            const float* __restrict__ b1, float* __restrict__ h1) {
  int idx = blockIdx.x * 256 + threadIdx.x;
  if (idx >= 16 * 2 * DD) return;
  int row = idx / (2 * DD), col = idx % (2 * DD);
  const float* a = y2 + (size_t)row * DD;
  float s = b1[col];
  for (int d = 0; d < DD; ++d) s += a[d] * W1[(size_t)d * (2 * DD) + col];
  float c = 0.7978845608028654f * (s + 0.044715f * s * s * s);   // tanh-gelu
  h1[idx] = 0.5f * s * (1.f + tanhf(c));
}

__global__ void ffn2_kernel(const float* __restrict__ h1, const float* __restrict__ W2,
                            const float* __restrict__ b2, float* __restrict__ tok) {
  int idx = blockIdx.x * 256 + threadIdx.x;
  if (idx >= 16 * DD) return;
  int row = idx / DD, col = idx % DD;
  const float* a = h1 + (size_t)row * (2 * DD);
  float s = b2[col];
  for (int d = 0; d < 2 * DD; ++d) s += a[d] * W2[(size_t)d * DD + col];
  int b = row >> 2, t = row & 3;
  tok[((size_t)(b * TT + t)) * DD + col] += s;
}

// ---------------- head: qtok mean, logits, softmax -> w (4x8) ----------------
__global__ void head_kernel(const float* __restrict__ tok, const float* __restrict__ mk,
                            float* __restrict__ wout) {
  __shared__ float logits[32];
  int idx = threadIdx.x;
  if (idx < 32) {
    int b = idx >> 3, m = idx & 7;
    float s = 0.f;
    for (int d = 0; d < DD; ++d) {
      float q = 0.25f * (tok[((size_t)(b * TT + 0)) * DD + d] + tok[((size_t)(b * TT + 1)) * DD + d] +
                         tok[((size_t)(b * TT + 2)) * DD + d] + tok[((size_t)(b * TT + 3)) * DD + d]);
      s += q * mk[(size_t)m * DD + d];
    }
    logits[idx] = s * 0.05103103630798288f;                    // 1/sqrt(384)
  }
  __syncthreads();
  if (idx < 4) {
    float m = -1e30f;
    for (int j = 0; j < 8; ++j) m = fmaxf(m, logits[idx * 8 + j]);
    float e[8], sum = 0.f;
    for (int j = 0; j < 8; ++j) { e[j] = __expf(logits[idx * 8 + j] - m); sum += e[j]; }
    float inv = 1.f / sum;
    for (int j = 0; j < 8; ++j) wout[idx * 8 + j] = e[j] * inv;
  }
}

// ---------------- streaming hard-concrete tail (bandwidth bound) ----------------
// 4 consecutive sites per thread: 8x b128 NT loads (u,la), 1x b128 (phi),
// 4x b128 NT stores (one per batch plane).
__global__ void tail_kernel(const float* __restrict__ u, const float* __restrict__ la,
                            const float* __restrict__ phi, const float* __restrict__ w,
                            float* __restrict__ out, size_t nloc, size_t out_off) {
  __shared__ float ws_[32];
  if (threadIdx.x < 32) ws_[threadIdx.x] = w[threadIdx.x];
  __syncthreads();
  size_t base = ((size_t)blockIdx.x * 256 + threadIdx.x) * 4;
  if (base >= nloc) return;
  f4 ph = __builtin_nontemporal_load((const f4*)(phi + base));
  f4 dv0 = {}, dv1 = {}, dv2 = {}, dv3 = {};
  const float invBeta = 1.5f;
#pragma unroll
  for (int j = 0; j < 4; ++j) {
    size_t idx = base + j;
    const f4* u4 = (const f4*)(u + idx * 8);
    const f4* l4 = (const f4*)(la + idx * 8);
    f4 a0 = __builtin_nontemporal_load(u4);
    f4 a1 = __builtin_nontemporal_load(u4 + 1);
    f4 b0 = __builtin_nontemporal_load(l4);
    f4 b1 = __builtin_nontemporal_load(l4 + 1);
    float um[8] = {a0.x, a0.y, a0.z, a0.w, a1.x, a1.y, a1.z, a1.w};
    float lm[8] = {b0.x, b0.y, b0.z, b0.w, b1.x, b1.y, b1.z, b1.w};
    float z[8];
#pragma unroll
    for (int m = 0; m < 8; ++m) {
      float uc = fminf(fmaxf(um[m], 1e-6f), 1.f - 1e-6f);
      float t  = (__logf(uc) - __logf(1.f - uc) + lm[m]) * invBeta;
      float s  = 1.f / (1.f + __expf(-t));
      z[m] = fminf(fmaxf(s * 1.2f - 0.1f, 0.f), 1.f);          // zeta-gamma=1.2, gamma=-0.1
    }
    float phj = ph[j];
    float d0 = 0, d1 = 0, d2 = 0, d3 = 0;
#pragma unroll
    for (int m = 0; m < 8; ++m) {
      float zm = z[m];
      d0 += zm * ws_[m]; d1 += zm * ws_[8 + m];
      d2 += zm * ws_[16 + m]; d3 += zm * ws_[24 + m];
    }
    dv0[j] = d0 * phj; dv1[j] = d1 * phj; dv2[j] = d2 * phj; dv3[j] = d3 * phj;
  }
  float* o = out + out_off + base;
  __builtin_nontemporal_store(dv0, (f4*)(o));
  __builtin_nontemporal_store(dv1, (f4*)(o + OUT_BSTRIDE));
  __builtin_nontemporal_store(dv2, (f4*)(o + 2 * OUT_BSTRIDE));
  __builtin_nontemporal_store(dv3, (f4*)(o + 3 * OUT_BSTRIDE));
}

// ---------------- host ----------------
extern "C" void kernel_launch(void* const* d_in, const int* in_sizes, int n_in,
                              void* d_out, int out_size, void* d_ws, size_t ws_size,
                              hipStream_t stream) {
  const float* x     = (const float*)d_in[0];
  const float* u_a   = (const float*)d_in[1];
  const float* u_b   = (const float*)d_in[2];
  const float* phi_a = (const float*)d_in[3];
  const float* phi_b = (const float*)d_in[4];
  const float* la_a  = (const float*)d_in[5];
  const float* la_b  = (const float*)d_in[6];
  const float* Wp    = (const float*)d_in[7];
  const float* bp    = (const float*)d_in[8];
  const float* skl   = (const float*)d_in[9];
  const float* cenc  = (const float*)d_in[10];
  const float* ln1g  = (const float*)d_in[11];
  const float* ln1b  = (const float*)d_in[12];
  const float* Wqkv  = (const float*)d_in[13];
  const float* bqkv  = (const float*)d_in[14];
  const float* Wo    = (const float*)d_in[15];
  const float* bo    = (const float*)d_in[16];
  const float* ln2g  = (const float*)d_in[17];
  const float* ln2b  = (const float*)d_in[18];
  const float* W1    = (const float*)d_in[19];
  const float* b1    = (const float*)d_in[20];
  const float* W2    = (const float*)d_in[21];
  const float* b2    = (const float*)d_in[22];
  const float* mk    = (const float*)d_in[23];
  float* out = (float*)d_out;

  float* ws  = (float*)d_ws;
  float* h    = ws;                       // 1024*384
  float* tok  = h   + 1024 * DD;          // 1040*384
  float* y    = tok + BB * TT * DD;       // 1040*384
  float* qkv  = y   + BB * TT * DD;       // 1040*1152
  float* o4   = qkv + (size_t)BB * TT * 3 * DD;  // 16*384
  float* y2   = o4  + 16 * DD;            // 16*384
  float* h1   = y2  + 16 * DD;            // 16*768
  float* wsm  = h1  + 16 * 2 * DD;        // 32

  // 1) h = x @ Wp + bp   (1024 x 384 x 384) — WMMA
  { int mTiles = 1024 / 16, bands = (mTiles + 7) / 8, nStrips = DD / 64;
    gemm_bf16_wmma<<<bands * nStrips, 256, 0, stream>>>(x, Wp, bp, h, 1024, DD, DD); }
  // 2) tok = [skills ; h + count_enc]
  assemble_tok<<<(BB * TT * DD + 255) / 256, 256, 0, stream>>>(h, skl, cenc, tok);
  // 3) y = LN1(tok)
  ln_kernel<<<BB * TT, 128, 0, stream>>>(tok, y, ln1g, ln1b, 0);
  // 4) qkv = y @ Wqkv + bqkv  (1040 x 1152 x 384) — WMMA
  { int mTiles = BB * TT / 16, bands = (mTiles + 7) / 8, nStrips = 3 * DD / 64;
    gemm_bf16_wmma<<<bands * nStrips, 256, 0, stream>>>(y, Wqkv, bqkv, qkv, BB * TT, 3 * DD, DD); }
  // 5) attention (queries = 4 skill tokens only)
  attn_kernel<<<BB * HH, 256, 0, stream>>>(qkv, o4);
  // 6) tok[:, :4] += o4 @ Wo + bo
  oproj_kernel<<<(16 * DD + 255) / 256, 256, 0, stream>>>(o4, Wo, bo, tok);
  // 7) y2 = LN2(tok[:, :4])
  ln_kernel<<<16, 128, 0, stream>>>(tok, y2, ln2g, ln2b, 1);
  // 8-9) FFN on hot rows
  ffn1_kernel<<<(16 * 2 * DD + 255) / 256, 256, 0, stream>>>(y2, W1, b1, h1);
  ffn2_kernel<<<(16 * DD + 255) / 256, 256, 0, stream>>>(h1, W2, b2, tok);
  // 10) w = softmax(mean(tok[:, :4]) @ mkT / sqrt(D))
  head_kernel<<<1, 64, 0, stream>>>(tok, mk, wsm);
  // 11) streaming tails: delta_a then delta_b (4 sites per thread)
  { int blocks = (int)((NLOC / 4 + 255) / 256);
    tail_kernel<<<blocks, 256, 0, stream>>>(u_a, la_a, phi_a, wsm, out, NLOC, 0);
    tail_kernel<<<blocks, 256, 0, stream>>>(u_b, la_b, phi_b, wsm, out, NLOC, NLOC); }
}
